// DCNModel_12816182411985
// MI455X (gfx1250) — matrix operand order repaired and checked by reading, not compiled, not run on hardware
//
#include <hip/hip_runtime.h>

// ---------------------------------------------------------------------------
// Problem constants (from reference): B=16384, DIM=1024, H1=500, H2=200, H3=200
// Padded: H1p=512, H2p=256, H3p=256.  f16 WMMA: 16x16 tiles, K-step 32.
// ---------------------------------------------------------------------------
#define BROWS   16384
#define DIM_K   1024
#define H1P     512
#define H2P     256
#define H3P     256

typedef __attribute__((ext_vector_type(16))) _Float16 v16h;
typedef __attribute__((ext_vector_type(8)))  float    v8f;

union Frag16 {               // 16 f16 elements = 8 VGPRs
    v16h      v;
    _Float16  h[16];
    unsigned  u[8];
    uint4     q[2];
};

union H2Pack { unsigned u; _Float16 h[2]; };

// ---------------------------------------------------------------------------
// Pre-pack a row-major f32 weight matrix (Ksrc x Nsrc) into the CDNA5 WMMA
// B-matrix fragment layout (32x16 f16 tiles), zero-padded to Ktiles*32 x
// Ntiles*16.  Word index = ((kt*Ntiles + nt)*32 + lane)*8 + v.
// B layout (16-bit, 32x16): half=lane>>4; VGPR v -> K = half*8 + (v>>2)*16 +
// (v&3)*2 (+1); N = lane & 15.
// ---------------------------------------------------------------------------
__global__ void prepack_b(const float* __restrict__ W, int Ksrc, int Nsrc,
                          int Ktiles, int Ntiles, unsigned* __restrict__ dst)
{
    int idx   = blockIdx.x * 256 + threadIdx.x;
    int total = Ktiles * Ntiles * 256;
    if (idx >= total) return;
    int v    = idx & 7;
    int lane = (idx >> 3) & 31;
    int tile = idx >> 8;
    int nt   = tile % Ntiles;
    int kt   = tile / Ntiles;
    int n    = nt * 16 + (lane & 15);
    int k    = kt * 32 + (lane >> 4) * 8 + (v >> 2) * 16 + (v & 3) * 2;
    float f0 = (k     < Ksrc && n < Nsrc) ? W[(size_t)k       * Nsrc + n] : 0.0f;
    float f1 = (k + 1 < Ksrc && n < Nsrc) ? W[(size_t)(k + 1) * Nsrc + n] : 0.0f;
    H2Pack p;
    p.h[0] = (_Float16)f0;
    p.h[1] = (_Float16)f1;
    dst[idx] = p.u;
}

// ---------------------------------------------------------------------------
// Fragment loaders.
//   A (16x32, 16-bit A-layout): lane m=lane&15, half=lane>>4; VGPR v holds
//   K = half*8 + (v>>2)*16 + (v&3)*2 (+1) — contiguous dwords in row-major.
//   B: pre-packed, 32B per lane per 32x16 tile, 4 tiles at 1KB immediates.
// ---------------------------------------------------------------------------
template <bool A_IS_F32>
__device__ __forceinline__ void load_a(const char* __restrict__ aptr, Frag16& a)
{
    if constexpr (A_IS_F32) {
        union F8 { float4 q[2]; float f[8]; } lo, hi;
        lo.q[0] = *(const float4*)(aptr);        // k: base+0..3
        lo.q[1] = *(const float4*)(aptr + 16);   // k: base+4..7
        hi.q[0] = *(const float4*)(aptr + 64);   // k: base+16..19
        hi.q[1] = *(const float4*)(aptr + 80);   // k: base+20..23
#pragma unroll
        for (int j = 0; j < 8; ++j) {
            a.h[j]     = (_Float16)lo.f[j];
            a.h[8 + j] = (_Float16)hi.f[j];
        }
    } else {
        a.q[0] = *(const uint4*)(aptr);          // k: base+0..7   (f16)
        a.q[1] = *(const uint4*)(aptr + 32);     // k: base+16..23 (f16)
    }
}

__device__ __forceinline__ void load_b(const unsigned* __restrict__ bptr, Frag16* b)
{
#pragma unroll
    for (int t = 0; t < 4; ++t) {
        b[t].q[0] = *(const uint4*)(bptr + t * 256);
        b[t].q[1] = *(const uint4*)(bptr + t * 256 + 4);
    }
}

#define WMMA8(A_, B_)                                                        \
    _Pragma("unroll")                                                        \
    for (int m_ = 0; m_ < 2; ++m_)                                           \
        _Pragma("unroll")                                                    \
        for (int t_ = 0; t_ < 4; ++t_)                                       \
            acc[m_][t_] = __builtin_amdgcn_wmma_f32_16x16x32_f16(            \
                false, (A_)[m_].v, false, (B_)[t_].v, (short)0,              \
                acc[m_][t_], false, false);

// ---------------------------------------------------------------------------
// WMMA GEMM:  C = relu(A(MxK) * Bpacked(KxN) + bias),  M=16384.
// 256 threads = 8 waves.  Block tile 128M x 128N, wave tile 32M x 64N
// (two A fragments share four B fragments -> 8 WMMA per K-step).
// Software-pipelined K loop, 2-deep register double buffering.
// ---------------------------------------------------------------------------
template <bool A_IS_F32, bool OUT_F16>
__global__ __launch_bounds__(256)
void gemm_wmma(const void* __restrict__ Av, int lda,
               const unsigned* __restrict__ Bp, int NtilesTot,
               const float* __restrict__ bias, int nsrc,
               void* __restrict__ Ov, int ldc, int Ktiles)
{
    const int lane = threadIdx.x & 31;
    const int w    = threadIdx.x >> 5;
    const int wm   = w >> 1;          // 0..3
    const int wn   = w & 1;           // 0..1
    const int hl   = lane >> 4;       // half
    const int m0   = blockIdx.y * 128 + wm * 32;
    const int nt0  = blockIdx.x * 8 + wn * 4;
    const int mrow = m0 + (lane & 15);

    v8f acc[2][4];
#pragma unroll
    for (int m = 0; m < 2; ++m)
#pragma unroll
        for (int t = 0; t < 4; ++t)
#pragma unroll
            for (int i = 0; i < 8; ++i) acc[m][t][i] = 0.0f;

    const size_t    esz     = A_IS_F32 ? 4 : 2;
    const char*     aptr    = (const char*)Av + ((size_t)mrow * lda + hl * 8) * esz;
    const size_t    aoff16  = (size_t)16 * lda * esz;   // +16 rows
    const size_t    astride = 32 * esz;                 // +1 K-tile
    const unsigned* bptr    = Bp + ((size_t)nt0 * 32 + lane) * 8;
    const size_t    bstride = (size_t)NtilesTot * 256;

    Frag16 a0[2], a1[2], b0[4], b1[4];
    load_a<A_IS_F32>(aptr,                    a0[0]);
    load_a<A_IS_F32>(aptr + aoff16,           a0[1]);
    load_b(bptr, b0);
    load_a<A_IS_F32>(aptr + astride,          a1[0]);
    load_a<A_IS_F32>(aptr + astride + aoff16, a1[1]);
    load_b(bptr + bstride, b1);

    // Ktiles is even and >= 2 for all three GEMMs.
    for (int kt = 0; kt < Ktiles - 2; kt += 2) {
        const char*     ap2 = aptr + 2 * astride;
        const unsigned* bp2 = bptr + 2 * bstride;
        WMMA8(a0, b0)
        load_a<A_IS_F32>(ap2,          a0[0]);
        load_a<A_IS_F32>(ap2 + aoff16, a0[1]);
        load_b(bp2, b0);
        __builtin_prefetch(ap2 + 2 * astride, 0, 3);   // speculative, safe
        WMMA8(a1, b1)
        load_a<A_IS_F32>(ap2 + astride,          a1[0]);
        load_a<A_IS_F32>(ap2 + astride + aoff16, a1[1]);
        load_b(bp2 + bstride, b1);
        aptr = ap2;
        bptr = bp2;
    }
    WMMA8(a0, b0)
    WMMA8(a1, b1)

    // Epilogue: bias + relu; padded columns forced to zero.
    // C/D layout: VGPR r -> row m0 + m*16 + hl*8 + r, col = nt*16 + (lane&15).
#pragma unroll
    for (int m = 0; m < 2; ++m) {
#pragma unroll
        for (int t = 0; t < 4; ++t) {
            const int n  = (nt0 + t) * 16 + (lane & 15);
            const float bv = (n < nsrc) ? bias[n] : 0.0f;
#pragma unroll
            for (int r = 0; r < 8; ++r) {
                const int row = m0 + m * 16 + hl * 8 + r;
                float val = acc[m][t][r] + bv;
                val = val > 0.0f ? val : 0.0f;
                if (n >= nsrc) val = 0.0f;
                if constexpr (OUT_F16) {
                    ((_Float16*)Ov)[(size_t)row * ldc + n] = (_Float16)val;
                } else {
                    ((float*)Ov)[(size_t)row * ldc + n] = val;
                }
            }
        }
    }
}

// ---------------------------------------------------------------------------
// Fused cross-network + head.  One wave (32 lanes) per row; 8 rows per block.
//   xsum = sum(x[row,:]);  per column c: e=x; 3x (e = xsum*e*cw + cb + x);
//   pred = sigmoid(dot(e, Wl[0:1024]) + dot(d3, Wl[1024:1224]) + bl)
// ---------------------------------------------------------------------------
__global__ __launch_bounds__(256)
void cross_head(const float* __restrict__ x, const float* __restrict__ d3,
                const float* __restrict__ Wl, const float* __restrict__ bl,
                const float* __restrict__ cw, const float* __restrict__ cb,
                float* __restrict__ out)
{
    const int lane = threadIdx.x & 31;
    const int row  = blockIdx.x * 8 + (threadIdx.x >> 5);
    const float* xr = x + (size_t)row * DIM_K;

    float xs[32];
    float s = 0.0f;
#pragma unroll
    for (int k = 0; k < 32; ++k) { xs[k] = xr[k * 32 + lane]; s += xs[k]; }
#pragma unroll
    for (int m = 16; m > 0; m >>= 1) s += __shfl_xor(s, m, 32);

    float wl[32];
#pragma unroll
    for (int k = 0; k < 32; ++k) wl[k] = Wl[k * 32 + lane];

    float dsum = 0.0f;
#pragma unroll
    for (int i = 0; i < 7; ++i) {
        const int c = i * 32 + lane;
        if (c < 200) dsum += d3[(size_t)row * H3P + c] * Wl[DIM_K + c];
    }
#pragma unroll
    for (int m = 16; m > 0; m >>= 1) dsum += __shfl_xor(dsum, m, 32);

    const float blv = bl[0];
#pragma unroll
    for (int task = 0; task < 2; ++task) {
        float acc = 0.0f;
#pragma unroll 4
        for (int k = 0; k < 32; ++k) {
            const int c = k * 32 + lane;
            const float xv = xs[k];
            float e = xv;
#pragma unroll
            for (int j = 0; j < 3; ++j) {
                const size_t o = ((size_t)task * 3 + j) * DIM_K + c;
                e = s * e * cw[o] + cb[o] + xv;
            }
            acc += e * wl[k];
        }
#pragma unroll
        for (int m = 16; m > 0; m >>= 1) acc += __shfl_xor(acc, m, 32);
        if (lane == 0) {
            const float z = acc + dsum + blv;
            out[(size_t)task * BROWS + row] = 1.0f / (1.0f + __expf(-z));
        }
    }
}

// ---------------------------------------------------------------------------
// Launch
// ---------------------------------------------------------------------------
extern "C" void kernel_launch(void* const* d_in, const int* in_sizes, int n_in,
                              void* d_out, int out_size, void* d_ws, size_t ws_size,
                              hipStream_t stream)
{
    const float* x  = (const float*)d_in[0];
    // d_in[1]=show_index, d_in[2]=st: unused by the reference
    const float* W1 = (const float*)d_in[3];
    const float* b1 = (const float*)d_in[4];
    const float* W2 = (const float*)d_in[5];
    const float* b2 = (const float*)d_in[6];
    const float* W3 = (const float*)d_in[7];
    const float* b3 = (const float*)d_in[8];
    const float* Wl = (const float*)d_in[9];
    const float* bl = (const float*)d_in[10];
    const float* cw = (const float*)d_in[11];
    const float* cb = (const float*)d_in[12];
    float* out = (float*)d_out;

    char* ws = (char*)d_ws;
    // Workspace layout (bytes):
    const size_t OFF_B1 = 0;                        // 32*32*256*4  = 1 MB
    const size_t OFF_B2 = OFF_B1 + (size_t)32 * 32 * 256 * 4;   // 256 KB
    const size_t OFF_B3 = OFF_B2 + (size_t)16 * 16 * 256 * 4;   // 128 KB
    const size_t OFF_D1 = OFF_B3 + (size_t)8 * 16 * 256 * 4;    // 16 MB (f16)
    const size_t OFF_D2 = OFF_D1 + (size_t)BROWS * H1P * 2;     // 8 MB  (f16)
    const size_t OFF_D3 = OFF_D2 + (size_t)BROWS * H2P * 2;     // 16 MB (f32)

    unsigned*  B1p = (unsigned*)(ws + OFF_B1);
    unsigned*  B2p = (unsigned*)(ws + OFF_B2);
    unsigned*  B3p = (unsigned*)(ws + OFF_B3);
    _Float16*  d1  = (_Float16*)(ws + OFF_D1);
    _Float16*  d2  = (_Float16*)(ws + OFF_D2);
    float*     d3  = (float*)(ws + OFF_D3);

    // Pack weights into WMMA B-fragment layout (f16, zero padded).
    prepack_b<<<(32 * 32 * 256) / 256, 256, 0, stream>>>(W1, DIM_K, 500, 32, 32, B1p);
    prepack_b<<<(16 * 16 * 256) / 256, 256, 0, stream>>>(W2, 500,   200, 16, 16, B2p);
    prepack_b<<<( 8 * 16 * 256) / 256, 256, 0, stream>>>(W3, 200,   200,  8, 16, B3p);

    // d1 = relu(x @ W1 + b1)      M=16384, K=1024, N=512 (500 valid)
    gemm_wmma<true,  true ><<<dim3(H1P / 128, BROWS / 128), 256, 0, stream>>>(
        x, DIM_K, B1p, 32, b1, 500, d1, H1P, DIM_K / 32);
    // d2 = relu(d1 @ W2 + b2)     K=512, N=256 (200 valid)
    gemm_wmma<false, true ><<<dim3(H2P / 128, BROWS / 128), 256, 0, stream>>>(
        d1, H1P, B2p, 16, b2, 200, d2, H2P, H1P / 32);
    // d3 = relu(d2 @ W3 + b3)     K=256, N=256 (200 valid), f32 output
    gemm_wmma<false, false><<<dim3(H3P / 128, BROWS / 128), 256, 0, stream>>>(
        d2, H2P, B3p, 16, b3, 200, d3, H3P, H2P / 32);

    // Cross network + sigmoid heads (both tasks).
    cross_head<<<BROWS / 8, 256, 0, stream>>>(x, d3, Wl, bl, cw, cb, out);
}
